// Attention_67078799229161
// MI455X (gfx1250) — matrix operand order repaired and checked by reading, not compiled, not run on hardware
//
#include <hip/hip_runtime.h>
#include <hip/hip_bf16.h>

// ---------------------------------------------------------------------------
// Problem constants (match reference): S=2048, DIM=1024, H=16, D=64
// ---------------------------------------------------------------------------
#define SEQ   2048
#define DIMN  1024
#define NHEAD 16
#define DHEAD 64

typedef __attribute__((ext_vector_type(16))) __bf16 v16bf;
typedef __attribute__((ext_vector_type(8)))  __bf16 v8bf;
typedef __attribute__((ext_vector_type(8)))  float  v8f;
typedef __attribute__((ext_vector_type(4)))  float  v4f;
typedef __attribute__((ext_vector_type(4)))  int    v4i;

#define AS1 __attribute__((address_space(1)))
#define AS3 __attribute__((address_space(3)))

#if defined(__has_builtin)
#if __has_builtin(__builtin_amdgcn_global_load_async_to_lds_b128)
#define USE_ASYNC_LDS 1
#endif
#endif

// 16-byte global -> LDS copy; async (ASYNCcnt) when available.
__device__ __forceinline__ void copy16_g2l(const __bf16* gsrc, __bf16* ldst) {
#ifdef USE_ASYNC_LDS
  __builtin_amdgcn_global_load_async_to_lds_b128(
      (AS1 v4i*)gsrc, (AS3 v4i*)ldst, 0, 0);
#else
  *(v8bf*)ldst = *(const v8bf*)gsrc;
#endif
}
__device__ __forceinline__ void wait_g2l() {
#ifdef USE_ASYNC_LDS
#if __has_builtin(__builtin_amdgcn_s_wait_asynccnt)
  __builtin_amdgcn_s_wait_asynccnt(0);
#else
  asm volatile("s_wait_asynccnt 0" ::: "memory");
#endif
#endif
}

// D = A(16x32 bf16) * B(32x16 bf16) + C(16x16 f32)
__device__ __forceinline__ v8f wmma_bf16(v16bf a, v16bf b, v8f c) {
  return __builtin_amdgcn_wmma_f32_16x16x32_bf16(false, a, false, b, (short)0, c,
                                                 false, false);
}

// A-fragment (16x32, bf16, row-major src, leading dim ld elements).
// Lane L: row m=L%16. elems 0..7 -> K = half*8 + j ; elems 8..15 -> 16+half*8+j
__device__ __forceinline__ v16bf load_A16(const __bf16* base, int ld, int lane) {
  const int m = lane & 15, hf = (lane >> 4) & 1;
  const __bf16* p = base + (size_t)m * ld;
  v8bf lo = *(const v8bf*)(p + hf * 8);
  v8bf hi = *(const v8bf*)(p + 16 + hf * 8);
  return __builtin_shufflevector(lo, hi, 0, 1, 2, 3, 4, 5, 6, 7,
                                 8, 9, 10, 11, 12, 13, 14, 15);
}

// B-fragment (32x16). Source "N x K row-major": row n contiguous over K.
// Lane L: col n=L%16. elems 0..7 -> K = half*8 + j ; elems 8..15 -> 16+half*8+j
__device__ __forceinline__ v16bf load_B16(const __bf16* base, int ld, int lane) {
  const int n = lane & 15, hf = (lane >> 4) & 1;
  const __bf16* p = base + (size_t)n * ld;
  v8bf lo = *(const v8bf*)(p + hf * 8);
  v8bf hi = *(const v8bf*)(p + 16 + hf * 8);
  return __builtin_shufflevector(lo, hi, 0, 1, 2, 3, 4, 5, 6, 7,
                                 8, 9, 10, 11, 12, 13, 14, 15);
}

__device__ __forceinline__ float redmax16(float v) {
  v = fmaxf(v, __shfl_xor(v, 1, 32));
  v = fmaxf(v, __shfl_xor(v, 2, 32));
  v = fmaxf(v, __shfl_xor(v, 4, 32));
  v = fmaxf(v, __shfl_xor(v, 8, 32));
  return v;
}
__device__ __forceinline__ float redsum16(float v) {
  v += __shfl_xor(v, 1, 32);
  v += __shfl_xor(v, 2, 32);
  v += __shfl_xor(v, 4, 32);
  v += __shfl_xor(v, 8, 32);
  return v;
}

// ---------------------------------------------------------------------------
// Kernel 0: f32 -> bf16 conversion (x, W_qkv, W_proj), 4 elems/thread
// ---------------------------------------------------------------------------
__global__ void cvt_f32_bf16(const float* __restrict__ src,
                             __bf16* __restrict__ dst, int n4) {
  int i = (blockIdx.x * blockDim.x + threadIdx.x);
  if (i < n4) {
    v4f v = *(const v4f*)(src + i * 4);
    __bf16* d = dst + i * 4;
    d[0] = (__bf16)v[0]; d[1] = (__bf16)v[1];
    d[2] = (__bf16)v[2]; d[3] = (__bf16)v[3];
  }
}

// ---------------------------------------------------------------------------
// Kernel 1: qkv = x @ W_qkv^T + b, then RoPE(q,k) (q also *= 1/sqrt(D)).
// One wave = one 32(M) x 64(N) tile: 8 WMMAs per k-step, B frags reused 2x.
// Outputs: Qb,Kb as bf16 [h][s][d]; V stored transposed Vt [h][d][s].
// ---------------------------------------------------------------------------
__global__ __launch_bounds__(256) void qkv_rope_kernel(
    const __bf16* __restrict__ xb, const __bf16* __restrict__ wqkvb,
    const float* __restrict__ bqkv, const float* __restrict__ sinp,
    const float* __restrict__ cosp, __bf16* __restrict__ Qb,
    __bf16* __restrict__ Kb, __bf16* __restrict__ Vt) {
  const int lane = threadIdx.x & 31;
  const int wg   = blockIdx.x * 8 + (threadIdx.x >> 5);
  const int strip_n = wg % 48;          // 3*DIM/64
  const int strip_m = wg / 48;          // SEQ/32
  const int n0 = strip_n * 64;
  const int m0 = strip_m * 32;

  v8f acc[8] = {};
  const __bf16* arow0 = xb + (size_t)m0 * DIMN;
  const __bf16* arow1 = arow0 + (size_t)16 * DIMN;
#pragma unroll 2
  for (int k0 = 0; k0 < DIMN; k0 += 32) {
    v16bf aF0 = load_A16(arow0 + k0, DIMN, lane);
    v16bf aF1 = load_A16(arow1 + k0, DIMN, lane);
#pragma unroll
    for (int a = 0; a < 4; ++a) {
      v16bf bF = load_B16(wqkvb + (size_t)(n0 + 16 * a) * DIMN + k0, DIMN, lane);
      acc[a]     = wmma_bf16(aF0, bF, acc[a]);
      acc[4 + a] = wmma_bf16(aF1, bF, acc[4 + a]);
    }
  }

  const int hf = lane >> 4, nlo = lane & 15;
  const int part = n0 >> 10;            // 0=q 1=k 2=v
  const int hh = (n0 & 1023) >> 6;      // head

#pragma unroll
  for (int j = 0; j < 2; ++j) {
    const int mb = m0 + 16 * j;
    if (part < 2) {
      __bf16* dst = (part == 0 ? Qb : Kb) + (size_t)hh * SEQ * DHEAD;
      const float qs = (part == 0) ? 0.125f : 1.0f;  // 1/sqrt(64) folded into Q
#pragma unroll
      for (int a = 0; a < 2; ++a) {
        const int dlo = 16 * a + nlo, dhi = dlo + 32;
        const float blo = bqkv[n0 + dlo], bhi = bqkv[n0 + dhi];
#pragma unroll
        for (int v = 0; v < 8; ++v) {
          const int row = mb + v + 8 * hf;
          float clo = acc[4 * j + a][v] + blo;
          float chi = acc[4 * j + a + 2][v] + bhi;
          float cl = cosp[row * DHEAD + dlo], sl = sinp[row * DHEAD + dlo];
          float ch = cosp[row * DHEAD + dhi], sh = sinp[row * DHEAD + dhi];
          float olo = (clo * cl - chi * sl) * qs;  // rotate_half: -x2 for d<32
          float ohi = (chi * ch + clo * sh) * qs;  //             +x1 for d>=32
          dst[(size_t)row * DHEAD + dlo] = (__bf16)olo;
          dst[(size_t)row * DHEAD + dhi] = (__bf16)ohi;
        }
      }
    } else {
#pragma unroll
      for (int a = 0; a < 4; ++a) {
        const int d = 16 * a + nlo;
        const float b = bqkv[n0 + d];
#pragma unroll
        for (int v = 0; v < 8; ++v) {
          const int row = mb + v + 8 * hf;
          Vt[((size_t)hh * DHEAD + d) * SEQ + row] = (__bf16)(acc[4 * j + a][v] + b);
        }
      }
    }
  }
}

// ---------------------------------------------------------------------------
// Kernel 2: flash attention. One wave = (head, 16-row query tile); all 8
// waves of a block share one head, so each 32-key K chunk (4KB) and V chunk
// (4KB) is staged into LDS once per block (async-to-LDS when available) and
// consumed by all 8 waves (8x global-traffic cut). Per chunk per wave:
// 4 WMMA (QK^T) + online softmax + P via LDS round-trip + 4 WMMA (P@V).
// ---------------------------------------------------------------------------
__global__ __launch_bounds__(256) void flash_attn_kernel(
    const __bf16* __restrict__ Qb, const __bf16* __restrict__ Kb,
    const __bf16* __restrict__ Vt, __bf16* __restrict__ Ob) {
  __shared__ __align__(16) float  ptile[8][16 * 32];
  __shared__ __align__(16) __bf16 Kc[32 * DHEAD];   // [k in chunk][d]
  __shared__ __align__(16) __bf16 Vc[DHEAD * 32];   // [d][k in chunk]
  const int tid = threadIdx.x;
  const int lane = tid & 31;
  const int warp = tid >> 5;
  const int wg = blockIdx.x * 8 + warp;
  const int h  = wg >> 7;               // constant within the block
  const int qt = wg & 127;
  const int m0 = qt * 16;
  const int hf = lane >> 4, nlo = lane & 15, m = lane & 15;

  const __bf16* Qh = Qb + (size_t)h * SEQ * DHEAD;
  const __bf16* Kh = Kb + (size_t)h * SEQ * DHEAD;
  const __bf16* Vh = Vt + (size_t)h * DHEAD * SEQ;
  float* lds = &ptile[warp][0];

  const v16bf qA0 = load_A16(Qh + (size_t)m0 * DHEAD, DHEAD, lane);
  const v16bf qA1 = load_A16(Qh + (size_t)m0 * DHEAD + 32, DHEAD, lane);

  float mrun[8], lrun[8];
#pragma unroll
  for (int v = 0; v < 8; ++v) { mrun[v] = -__builtin_inff(); lrun[v] = 0.f; }
  v8f o[4] = {};

  // cooperative copy mapping: K flat (chunk is contiguous 4KB in Kh);
  // V: thread t -> row d=t/4, 16B piece p=t%4 of the 64B chunk row.
  const int vd = tid >> 2, vp = tid & 3;

  for (int kc = 0; kc < SEQ; kc += 32) {
    __syncthreads();                    // previous chunk fully consumed
    copy16_g2l(Kh + (size_t)kc * DHEAD + tid * 8, Kc + tid * 8);
    copy16_g2l(Vh + (size_t)vd * SEQ + kc + vp * 8, Vc + vd * 32 + vp * 8);
    if (kc + 32 < SEQ) {                // warm next chunk (global_prefetch_b8)
      __builtin_prefetch(Kh + (size_t)(kc + 32) * DHEAD + tid * 8, 0, 1);
      __builtin_prefetch(Vh + (size_t)vd * SEQ + kc + 32 + vp * 8, 0, 1);
    }
    wait_g2l();
    __syncthreads();                    // chunk visible to all waves

    // scores: two 16x16 tiles over keys [kc,kc+16) and [kc+16,kc+32)
    v8f s0 = {}, s1 = {};
    s0 = wmma_bf16(qA0, load_B16(Kc, DHEAD, lane), s0);
    s0 = wmma_bf16(qA1, load_B16(Kc + 32, DHEAD, lane), s0);
    s1 = wmma_bf16(qA0, load_B16(Kc + 16 * DHEAD, DHEAD, lane), s1);
    s1 = wmma_bf16(qA1, load_B16(Kc + 16 * DHEAD + 32, DHEAD, lane), s1);

    // online softmax; row m = v + 8*half lives across the 16 lanes of a half
    float scale[8];
#pragma unroll
    for (int v = 0; v < 8; ++v) {
      float mx = redmax16(fmaxf(s0[v], s1[v]));
      float mnew = fmaxf(mrun[v], mx);
      scale[v] = __expf(mrun[v] - mnew);
      float p0 = __expf(s0[v] - mnew);
      float p1 = __expf(s1[v] - mnew);
      lrun[v] = lrun[v] * scale[v] + redsum16(p0 + p1);
      mrun[v] = mnew;
      lds[(v + 8 * hf) * 32 + nlo] = p0;         // C-layout -> 16x32 tile
      lds[(v + 8 * hf) * 32 + 16 + nlo] = p1;
    }
#pragma unroll
    for (int a = 0; a < 4; ++a)
#pragma unroll
      for (int v = 0; v < 8; ++v) o[a][v] *= scale[v];

    asm volatile("s_wait_dscnt 0" ::: "memory");

    // reload P as A-fragment (f32 -> bf16); wave-private tile, no barrier
    const float* prow = lds + m * 32;
    v4f f0 = *(const v4f*)(prow + hf * 8);
    v4f f1 = *(const v4f*)(prow + hf * 8 + 4);
    v4f f2 = *(const v4f*)(prow + 16 + hf * 8);
    v4f f3 = *(const v4f*)(prow + 16 + hf * 8 + 4);
    v16bf pA;
#pragma unroll
    for (int i = 0; i < 4; ++i) {
      pA[i]      = (__bf16)f0[i];
      pA[4 + i]  = (__bf16)f1[i];
      pA[8 + i]  = (__bf16)f2[i];
      pA[12 + i] = (__bf16)f3[i];
    }
    // O += P (16x32) @ V (32x64); Vc rows (d) contiguous over k
#pragma unroll
    for (int a = 0; a < 4; ++a)
      o[a] = wmma_bf16(pA, load_B16(Vc + (size_t)(16 * a) * 32, 32, lane), o[a]);
  }

  // normalize and store to Ob[s][h*64 + d] (bf16)
#pragma unroll
  for (int v = 0; v < 8; ++v) {
    const float inv = 1.0f / lrun[v];
    const int row = m0 + v + 8 * hf;
#pragma unroll
    for (int a = 0; a < 4; ++a)
      Ob[(size_t)row * DIMN + h * DHEAD + 16 * a + nlo] = (__bf16)(o[a][v] * inv);
  }
}

// ---------------------------------------------------------------------------
// Kernel 3: out = Ob @ W_proj^T + b_proj (f32 out), 32x64 tile per wave
// ---------------------------------------------------------------------------
__global__ __launch_bounds__(256) void proj_kernel(
    const __bf16* __restrict__ Ob, const __bf16* __restrict__ wprojb,
    const float* __restrict__ bproj, float* __restrict__ out) {
  const int lane = threadIdx.x & 31;
  const int wg = blockIdx.x * 8 + (threadIdx.x >> 5);
  const int strip_n = wg & 15;          // DIM/64
  const int strip_m = wg >> 4;          // SEQ/32
  const int n0 = strip_n * 64, m0 = strip_m * 32;

  v8f acc[8] = {};
  const __bf16* arow0 = Ob + (size_t)m0 * DIMN;
  const __bf16* arow1 = arow0 + (size_t)16 * DIMN;
#pragma unroll 2
  for (int k0 = 0; k0 < DIMN; k0 += 32) {
    v16bf aF0 = load_A16(arow0 + k0, DIMN, lane);
    v16bf aF1 = load_A16(arow1 + k0, DIMN, lane);
#pragma unroll
    for (int a = 0; a < 4; ++a) {
      v16bf bF = load_B16(wprojb + (size_t)(n0 + 16 * a) * DIMN + k0, DIMN, lane);
      acc[a]     = wmma_bf16(aF0, bF, acc[a]);
      acc[4 + a] = wmma_bf16(aF1, bF, acc[4 + a]);
    }
  }
  const int hf = lane >> 4, nlo = lane & 15;
#pragma unroll
  for (int j = 0; j < 2; ++j) {
#pragma unroll
    for (int a = 0; a < 4; ++a) {
      const int col = n0 + 16 * a + nlo;
      const float b = bproj[col];
#pragma unroll
      for (int v = 0; v < 8; ++v) {
        const int row = m0 + 16 * j + v + 8 * hf;
        out[(size_t)row * DIMN + col] = acc[4 * j + a][v] + b;
      }
    }
  }
}

// ---------------------------------------------------------------------------
extern "C" void kernel_launch(void* const* d_in, const int* in_sizes, int n_in,
                              void* d_out, int out_size, void* d_ws,
                              size_t ws_size, hipStream_t stream) {
  const float* x     = (const float*)d_in[0];
  const float* sinp  = (const float*)d_in[1];
  const float* cosp  = (const float*)d_in[2];
  const float* Wqkv  = (const float*)d_in[3];
  const float* bqkv  = (const float*)d_in[4];
  const float* Wproj = (const float*)d_in[5];
  const float* bproj = (const float*)d_in[6];
  float* out = (float*)d_out;

  // workspace carve-up (bf16 elements): 28 MiB total
  __bf16* xb     = (__bf16*)d_ws;                 // 2048*1024
  __bf16* wqkvb  = xb + (size_t)SEQ * DIMN;       // 3072*1024
  __bf16* wprojb = wqkvb + (size_t)3 * DIMN * DIMN;
  __bf16* Qb     = wprojb + (size_t)DIMN * DIMN;  // [h][s][d]
  __bf16* Kb     = Qb + (size_t)NHEAD * SEQ * DHEAD;
  __bf16* Vt     = Kb + (size_t)NHEAD * SEQ * DHEAD;  // [h][d][s]
  __bf16* Ob     = Vt + (size_t)NHEAD * SEQ * DHEAD;  // [s][dim]

  {
    int n4 = (SEQ * DIMN) / 4;
    cvt_f32_bf16<<<(n4 + 255) / 256, 256, 0, stream>>>(x, xb, n4);
    n4 = (3 * DIMN * DIMN) / 4;
    cvt_f32_bf16<<<(n4 + 255) / 256, 256, 0, stream>>>(Wqkv, wqkvb, n4);
    n4 = (DIMN * DIMN) / 4;
    cvt_f32_bf16<<<(n4 + 255) / 256, 256, 0, stream>>>(Wproj, wprojb, n4);
  }

  // (SEQ/32)*(3*DIM/64) = 64*48 = 3072 waves, 8 per block
  qkv_rope_kernel<<<384, 256, 0, stream>>>(xb, wqkvb, bqkv, sinp, cosp,
                                           Qb, Kb, Vt);
  // 16 heads * 128 q-tiles = 2048 waves
  flash_attn_kernel<<<256, 256, 0, stream>>>(Qb, Kb, Vt, Ob);
  // (SEQ/32)*(DIM/64) = 64*16 = 1024 waves
  proj_kernel<<<128, 256, 0, stream>>>(Ob, wprojb, bproj, out);
}
